// DecLayer_77970836291590
// MI455X (gfx1250) — matrix operand order repaired and checked by reading, not compiled, not run on hardware
//
#include <hip/hip_runtime.h>
#include <hip/hip_bf16.h>
#include <math.h>

#define H     128
#define NIN   384
#define FDIM  512        // H + NIN
#define KNB   48
#define BN    8192       // B*N
#define SCALE 30.0f
#define EPS   1e-5f
#define AROW  520        // 512 + 8 halfword pad  (16B/row bank rotation)
#define MROW  136        // 128 + 8
#define A2ROW 520        // 512 + 8

typedef __attribute__((ext_vector_type(16))) __bf16        v16bf;
typedef __attribute__((ext_vector_type(8)))  __bf16        v8bf;
typedef __attribute__((ext_vector_type(8)))  float         v8f;
typedef __attribute__((ext_vector_type(4)))  float         f4v;
typedef __attribute__((ext_vector_type(4)))  unsigned short us4;

__device__ __forceinline__ unsigned short f2bf(float x) {
  unsigned int u = __builtin_bit_cast(unsigned int, x);
  u += 0x7FFFu + ((u >> 16) & 1u);            // round-to-nearest-even
  return (unsigned short)(u >> 16);
}

// Branch-free erf (Abramowitz-Stegun 7.1.26, |err| < 1.5e-7): FMA chain +
// hardware v_rcp_f32 / v_exp_f32 only -- no libdevice exec-mask branches.
__device__ __forceinline__ float erf_bf(float x) {
  float ax = __builtin_fabsf(x);
  float t  = __builtin_amdgcn_rcpf(__builtin_fmaf(0.3275911f, ax, 1.0f));
  float p  = __builtin_fmaf(1.061405429f, t, -1.453152027f);
  p = __builtin_fmaf(p, t, 1.421413741f);
  p = __builtin_fmaf(p, t, -0.284496736f);
  p = __builtin_fmaf(p, t, 0.254829592f);
  p *= t;
  float e  = __expf(-ax * ax);                 // v_exp_f32
  float r  = __builtin_fmaf(-p, e, 1.0f);
  return __builtin_copysignf(r, x);
}
__device__ __forceinline__ float gelu_f(float x) {
  return 0.5f * x * (1.0f + erf_bf(x * 0.70710678118654752440f));
}

// A fragment (16x32 bf16, MxK) per ISA layout: lanes 0-15 row M=lane, K {0..7,16..23};
// lanes 16-31 row M=lane-16, K {8..15,24..31}.  Two 16B LDS reads.
__device__ __forceinline__ v16bf load_afrag(const unsigned short* a, int rowStride,
                                            int mt, int kb, int lane) {
  const int r   = lane & 15;
  const int hi  = lane >> 4;
  const int row = mt * 16 + r;
  const int c0  = kb * 32 + hi * 8;
  v8bf lo  = *reinterpret_cast<const v8bf*>(a + row * rowStride + c0);
  v8bf hi8 = *reinterpret_cast<const v8bf*>(a + row * rowStride + c0 + 16);
  v16bf out;
#pragma unroll
  for (int i = 0; i < 8; ++i) { out[i] = lo[i]; out[i + 8] = hi8[i]; }
  return out;
}

// B fragment: weights pre-packed tile-major [(nt*numKt+kt)*32 + lane]*16 halfwords,
// so each lane issues one 32B global load (L2-resident weights).
__device__ __forceinline__ v16bf load_bfrag(const unsigned short* __restrict__ wbf,
                                            int numKt, int nt, int kt, int lane) {
  return *reinterpret_cast<const v16bf*>(wbf + (((nt * numKt + kt) * 32 + lane) << 4));
}

__device__ __forceinline__ v8f wmma_bf16(v16bf a, v16bf b, v8f c) {
  return __builtin_amdgcn_wmma_f32_16x16x32_bf16(false, a, false, b, (short)0, c,
                                                 false, false);
}

// one 48xK @ KxN(=128) MLP layer; wave w owns N-tile nt=w; gelu(acc + bias) -> Mdst
__device__ __forceinline__ void mlp_layer3(const unsigned short* Asrc, int arow, int numKb,
                                           const unsigned short* __restrict__ Wbf,
                                           const float* __restrict__ bias,
                                           unsigned short* Mdst, int w, int lane) {
  const int nt = w;
  v8f z = {0.f, 0.f, 0.f, 0.f, 0.f, 0.f, 0.f, 0.f};
  v8f acc0 = z, acc1 = z, acc2 = z;
  for (int kb = 0; kb < numKb; ++kb) {
    v16bf b = load_bfrag(Wbf, numKb, nt, kb, lane);
    acc0 = wmma_bf16(load_afrag(Asrc, arow, 0, kb, lane), b, acc0);
    acc1 = wmma_bf16(load_afrag(Asrc, arow, 1, kb, lane), b, acc1);
    acc2 = wmma_bf16(load_afrag(Asrc, arow, 2, kb, lane), b, acc2);
  }
  const int   col = nt * 16 + (lane & 15);
  const float bc  = bias[col];
  const int   rb  = (lane >> 4) << 3;            // D layout: rows rb..rb+7 of tile
  v8f accs[3] = {acc0, acc1, acc2};
#pragma unroll
  for (int mt = 0; mt < 3; ++mt)
#pragma unroll
    for (int j = 0; j < 8; ++j)
      Mdst[(mt * 16 + rb + j) * MROW + col] = f2bf(gelu_f(accs[mt][j] + bc));
}

// ---------------- kernel 1: per-node message MLP + masked neighbor reduction ----------
__global__ __launch_bounds__(256) void k_message(
    const float* __restrict__ hV, const float* __restrict__ hE,
    const float* __restrict__ maskA,
    const unsigned short* __restrict__ W1bf, const float* __restrict__ b1,
    const unsigned short* __restrict__ W2bf, const float* __restrict__ b2,
    const unsigned short* __restrict__ W3bf, const float* __restrict__ b3,
    float* __restrict__ dhOut) {
  __shared__ __align__(16) unsigned short Abuf[KNB * AROW];  // 48x512 bf16 (padded); reused as M2
  __shared__ __align__(16) unsigned short M1[KNB * MROW];
  __shared__ float maskLds[KNB];
  unsigned short* M2 = Abuf;   // Abuf dead after layer 1; alias to stay < 64KB LDS

  const int node = blockIdx.x;
  const int tid  = threadIdx.x;
  const int lane = tid & 31;
  const int w    = tid >> 5;

  // stage A = [h_V (bcast) || h_E] as bf16
  const float* __restrict__ hVrow = hV + (size_t)node * H;
  for (int i = tid; i < KNB * H; i += 256) {
    int r = i >> 7, c = i & (H - 1);
    Abuf[r * AROW + c] = f2bf(hVrow[c]);
  }
  // h_E: 603MB streamed exactly once -> non-temporal so it can't evict hot weights in L2
  const float* __restrict__ hErow = hE + (size_t)node * (KNB * NIN);
  for (int i = tid; i < (KNB * NIN) / 4; i += 256) {
    f4v v = __builtin_nontemporal_load(reinterpret_cast<const f4v*>(hErow + i * 4));
    int e = i * 4;
    int r = e / NIN, c = e - r * NIN;
    us4 p; p[0] = f2bf(v[0]); p[1] = f2bf(v[1]); p[2] = f2bf(v[2]); p[3] = f2bf(v[3]);
    *reinterpret_cast<us4*>(&Abuf[r * AROW + H + c]) = p;
  }
  if (tid < KNB) maskLds[tid] = maskA[(size_t)node * KNB + tid];
  __syncthreads();

  mlp_layer3(Abuf, AROW, FDIM / 32, W1bf, b1, M1, w, lane);   // 48x512 @ 512x128
  __syncthreads();
  mlp_layer3(M1, MROW, H / 32, W2bf, b2, M2, w, lane);        // 48x128 @ 128x128
  __syncthreads();

  // layer 3 (no gelu) + masked sum over K neighbors, fused in registers
  {
    const int nt = w;
    v8f z = {0.f, 0.f, 0.f, 0.f, 0.f, 0.f, 0.f, 0.f};
    v8f acc0 = z, acc1 = z, acc2 = z;
    for (int kb = 0; kb < H / 32; ++kb) {
      v16bf b = load_bfrag(W3bf, H / 32, nt, kb, lane);
      acc0 = wmma_bf16(load_afrag(M2, MROW, 0, kb, lane), b, acc0);
      acc1 = wmma_bf16(load_afrag(M2, MROW, 1, kb, lane), b, acc1);
      acc2 = wmma_bf16(load_afrag(M2, MROW, 2, kb, lane), b, acc2);
    }
    const int   col = nt * 16 + (lane & 15);
    const float bc  = b3[col];
    const int   rb  = (lane >> 4) << 3;
    v8f accs[3] = {acc0, acc1, acc2};
    float s = 0.f;
#pragma unroll
    for (int mt = 0; mt < 3; ++mt)
#pragma unroll
      for (int j = 0; j < 8; ++j)
        s += maskLds[mt * 16 + rb + j] * (accs[mt][j] + bc);
    s += __shfl_xor(s, 16, 32);                 // combine row-halves (lanes L / L+16)
    if (lane < 16) dhOut[(size_t)node * H + col] = s * (1.0f / SCALE);
  }
}

// ---------------- kernel 2: LN1 -> FFN (WMMA) -> LN2 -> node mask ---------------------
__global__ __launch_bounds__(256) void k_ffn(
    const float* __restrict__ hV, const float* __restrict__ maskV,
    const float* __restrict__ g1, const float* __restrict__ be1,
    const float* __restrict__ g2, const float* __restrict__ be2,
    const unsigned short* __restrict__ Wibf, const float* __restrict__ bi,
    const unsigned short* __restrict__ Wobf, const float* __restrict__ bo,
    const float* __restrict__ alphaP, const float* __restrict__ dhIn,
    float* __restrict__ out) {
  __shared__ float xbuf[16 * H];
  __shared__ float hbuf[16 * H];
  __shared__ __align__(16) unsigned short Ab[16 * MROW];
  __shared__ __align__(16) unsigned short A2[16 * A2ROW];

  const int tid = threadIdx.x, lane = tid & 31, w = tid >> 5;
  const int node0 = blockIdx.x * 16;
  const float alpha = alphaP[0];

  for (int i = tid; i < 16 * H; i += 256) {
    size_t g = (size_t)node0 * H + i;
    xbuf[i] = alpha * dhIn[g] + hV[g];          // rezero residual into LN1
  }
  __syncthreads();

  // LN1: wave w handles rows 2w,2w+1; 4 elems/lane + shuffle tree
  for (int rr = 0; rr < 2; ++rr) {
    int r = w * 2 + rr;
    f4v v = *reinterpret_cast<const f4v*>(&xbuf[r * H + lane * 4]);
    float s = v[0] + v[1] + v[2] + v[3];
    float q = v[0] * v[0] + v[1] * v[1] + v[2] * v[2] + v[3] * v[3];
#pragma unroll
    for (int off = 16; off > 0; off >>= 1) { s += __shfl_xor(s, off, 32); q += __shfl_xor(q, off, 32); }
    float mu = s * (1.0f / H);
    float inv = rsqrtf(q * (1.0f / H) - mu * mu + EPS);
#pragma unroll
    for (int j = 0; j < 4; ++j) {
      int c = lane * 4 + j;
      float xn = (v[j] - mu) * inv * g1[c] + be1[c];
      hbuf[r * H + c] = xn;                      // keep fp32 residual
      Ab[r * MROW + c] = f2bf(xn);               // bf16 A for FFN
    }
  }
  __syncthreads();

  // FFN1: 16x128 @ 128x512 -> gelu -> A2
  for (int i = 0; i < 4; ++i) {
    int nt = w + 8 * i;
    v8f acc = {0.f, 0.f, 0.f, 0.f, 0.f, 0.f, 0.f, 0.f};
    for (int kb = 0; kb < H / 32; ++kb)
      acc = wmma_bf16(load_afrag(Ab, MROW, 0, kb, lane),
                      load_bfrag(Wibf, H / 32, nt, kb, lane), acc);
    int col = nt * 16 + (lane & 15);
    float bc = bi[col];
    int rb = (lane >> 4) << 3;
#pragma unroll
    for (int j = 0; j < 8; ++j)
      A2[(rb + j) * A2ROW + col] = f2bf(gelu_f(acc[j] + bc));
  }
  __syncthreads();

  // FFN2: 16x512 @ 512x128 -> rezero residual into xbuf
  {
    int nt = w;
    v8f acc = {0.f, 0.f, 0.f, 0.f, 0.f, 0.f, 0.f, 0.f};
    for (int kb = 0; kb < (4 * H) / 32; ++kb)
      acc = wmma_bf16(load_afrag(A2, A2ROW, 0, kb, lane),
                      load_bfrag(Wobf, (4 * H) / 32, nt, kb, lane), acc);
    int col = nt * 16 + (lane & 15);
    float bc = bo[col];
    int rb = (lane >> 4) << 3;
#pragma unroll
    for (int j = 0; j < 8; ++j)
      xbuf[(rb + j) * H + col] = alpha * (acc[j] + bc) + hbuf[(rb + j) * H + col];
  }
  __syncthreads();

  // LN2 + node mask -> out (write-once: non-temporal store)
  for (int rr = 0; rr < 2; ++rr) {
    int r = w * 2 + rr;
    f4v v = *reinterpret_cast<const f4v*>(&xbuf[r * H + lane * 4]);
    float s = v[0] + v[1] + v[2] + v[3];
    float q = v[0] * v[0] + v[1] * v[1] + v[2] * v[2] + v[3] * v[3];
#pragma unroll
    for (int off = 16; off > 0; off >>= 1) { s += __shfl_xor(s, off, 32); q += __shfl_xor(q, off, 32); }
    float mu = s * (1.0f / H);
    float inv = rsqrtf(q * (1.0f / H) - mu * mu + EPS);
    float mv = maskV[node0 + r];
    f4v o;
#pragma unroll
    for (int j = 0; j < 4; ++j) {
      int c = lane * 4 + j;
      o[j] = ((v[j] - mu) * inv * g2[c] + be2[c]) * mv;
    }
    __builtin_nontemporal_store(o, reinterpret_cast<f4v*>(out + (size_t)(node0 + r) * H + lane * 4));
  }
}

// ---------------- weight pre-pack: fp32 [Kd,Nd] -> bf16 WMMA B-fragment layout --------
__global__ void k_pack(const float* __restrict__ W, unsigned short* __restrict__ out,
                       int Kd, int Nd) {
  int t = blockIdx.x * blockDim.x + threadIdx.x;
  if (t >= Kd * Nd) return;
  int k = t / Nd, n = t - k * Nd;
  int kt = k >> 5, kr = k & 31, nt = n >> 4, nr = n & 15;
  int lane = ((kr >> 3) & 1) ? (16 + nr) : nr; // B layout: lanes 0-15 K {0..7,16..23}, 16-31 K {8..15,24..31}
  int h = (kr & 7) + ((kr >> 4) << 3);
  int off = ((nt * (Kd >> 5) + kt) * 32 + lane) * 16 + h;
  out[off] = f2bf(W[t]);
}

extern "C" void kernel_launch(void* const* d_in, const int* in_sizes, int n_in,
                              void* d_out, int out_size, void* d_ws, size_t ws_size,
                              hipStream_t stream) {
  (void)in_sizes; (void)n_in; (void)out_size; (void)ws_size;
  const float* hV    = (const float*)d_in[0];
  const float* hE    = (const float*)d_in[1];
  const float* maskV = (const float*)d_in[2];
  const float* maskA = (const float*)d_in[3];
  const float* W1 = (const float*)d_in[4];  const float* b1 = (const float*)d_in[5];
  const float* W2 = (const float*)d_in[6];  const float* b2 = (const float*)d_in[7];
  const float* W3 = (const float*)d_in[8];  const float* b3 = (const float*)d_in[9];
  const float* g1 = (const float*)d_in[10]; const float* be1 = (const float*)d_in[11];
  const float* g2 = (const float*)d_in[12]; const float* be2 = (const float*)d_in[13];
  const float* Wi = (const float*)d_in[14]; const float* bi = (const float*)d_in[15];
  const float* Wo = (const float*)d_in[16]; const float* bo = (const float*)d_in[17];
  const float* alpha = (const float*)d_in[18];
  float* out = (float*)d_out;

  char* ws = (char*)d_ws;
  const size_t OFF_W1 = 0;
  const size_t OFF_W2 = OFF_W1 + (size_t)FDIM * H * 2;   // 131072
  const size_t OFF_W3 = OFF_W2 + (size_t)H * H * 2;      // +32768
  const size_t OFF_WI = OFF_W3 + (size_t)H * H * 2;
  const size_t OFF_WO = OFF_WI + (size_t)H * (4 * H) * 2;
  const size_t OFF_DH = OFF_WO + (size_t)(4 * H) * H * 2;
  unsigned short* W1bf = (unsigned short*)(ws + OFF_W1);
  unsigned short* W2bf = (unsigned short*)(ws + OFF_W2);
  unsigned short* W3bf = (unsigned short*)(ws + OFF_W3);
  unsigned short* Wibf = (unsigned short*)(ws + OFF_WI);
  unsigned short* Wobf = (unsigned short*)(ws + OFF_WO);
  float*          dhWs = (float*)(ws + OFF_DH);

  k_pack<<<(FDIM * H + 255) / 256, 256, 0, stream>>>(W1, W1bf, FDIM, H);
  k_pack<<<(H * H + 255) / 256, 256, 0, stream>>>(W2, W2bf, H, H);
  k_pack<<<(H * H + 255) / 256, 256, 0, stream>>>(W3, W3bf, H, H);
  k_pack<<<(H * 4 * H + 255) / 256, 256, 0, stream>>>(Wi, Wibf, H, 4 * H);
  k_pack<<<(4 * H * H + 255) / 256, 256, 0, stream>>>(Wo, Wobf, 4 * H, H);

  k_message<<<BN, 256, 0, stream>>>(hV, hE, maskA, W1bf, b1, W2bf, b2, W3bf, b3, dhWs);
  k_ffn<<<BN / 16, 256, 0, stream>>>(hV, maskV, g1, be1, g2, be2,
                                     Wibf, bi, Wobf, bo, alpha, dhWs, out);
}
